// SimpleSocialLSTM_64321430225602
// MI455X (gfx1250) — matrix operand order repaired and checked by reading, not compiled, not run on hardware
//
#include <hip/hip_runtime.h>
#include <hip/hip_bf16.h>

typedef __attribute__((ext_vector_type(16))) _Float16 v16h;
typedef __attribute__((ext_vector_type(8)))  _Float16 v8h;
typedef __attribute__((ext_vector_type(8)))  float    v8f;

namespace {
constexpr int kH    = 256;   // hidden
constexpr int kT    = 30;    // encoder timesteps
constexpr int kIn   = 7;     // encoder input features
constexpr int kK    = 8;     // neighbors
constexpr int kPred = 25;    // decoder steps
constexpr int kB    = 8192;  // batch

// ---- encoder: extended-K recurrence ---------------------------------------
// K = 256 (h) + 7 (x features) + 1 (bias column) + 24 zero pad = 288 = 9 tiles
constexpr int kKtE  = 9;     // encoder k-tiles
constexpr int kLdaE = 296;   // hA row stride (halves): 592B, conflict-free
constexpr int kWBHalvesE = 16 * 2 * kKtE * 512;              // 147456
constexpr size_t kWBBytesE = (size_t)kWBHalvesE * 2;         // 294912
constexpr size_t kHABytesE = (size_t)16 * kLdaE * 2;         // 9472
constexpr size_t kXsBytes  = (size_t)16 * kT * kIn * 4;      // 13440
constexpr size_t kEncLds   = kWBBytesE + kHABytesE + kXsBytes; // 317824 <= 320K

// ---- decoder: plain K=256 recurrence --------------------------------------
constexpr int kLda  = 264;   // decoder hA row stride (halves)
constexpr int kWBHalves = 16 * 2 * 8 * 512;                  // 131072
constexpr size_t kWBBytes = (size_t)kWBHalves * 2;           // 262144
constexpr size_t kHABytes = (size_t)16 * kLda * 2;           // 8448
constexpr size_t kDecLds  = kWBBytes + kHABytes + 2048 + 256; // 272896

__device__ __forceinline__ float sigf(float x) {
  return 1.0f / (1.0f + __expf(-x));
}
__device__ __forceinline__ float tanh_fast(float x) {
  // 1 - 2/(e^{2x}+1): saturates cleanly to +-1, no inf/inf NaN
  float e = __expf(2.0f * x);
  return 1.0f - 2.0f / (e + 1.0f);
}
} // namespace

// ---------------------------------------------------------------------------
// Fused LSTM encoder, extended-K formulation. One workgroup = 16 batch rows,
// 512 threads = 16 waves; wave s owns hidden columns [s*16, s*16+16).
// The A operand per step is [h_{t-1} (256) | x_t (7) | 1.0 | zeros] so the
// input projection and bias ride inside the same WMMAs as the recurrence:
//   gates {i, f}: 9 weight fragments each, register-resident (144 VGPRs)
//   gates {g, o}: fragments pre-formatted into dynamic LDS (288 KB), ds_load'ed
// Accumulators start at zero; no per-step VALU projection work remains.
// ---------------------------------------------------------------------------
__global__ __launch_bounds__(512, 1)
void lstm_encoder_kernel(const float* __restrict__ x,     // [Bt, kT, kIn]
                         const float* __restrict__ Wih,   // [4H, kIn]
                         const float* __restrict__ Whh,   // [4H, H]
                         const float* __restrict__ bih,   // [4H]
                         const float* __restrict__ bhh,   // [4H]
                         float* __restrict__ hout)        // [Bt, H]
{
  extern __shared__ char smem[];
  _Float16* wB = (_Float16*)(smem);                        // 288 KB
  _Float16* hA = (_Float16*)(smem + kWBBytesE);            // 9472 B
  float*    xs = (float*)(smem + kWBBytesE + kHABytesE);   // 13440 B

  const int  tid    = threadIdx.x;
  const int  lane   = tid & 31;
  const int  s      = tid >> 5;          // wave = hidden slice 0..15
  const int  j0     = s * 16;            // hidden slice base
  const int  nIdx   = lane & 15;         // N (col) / M (row) index within tile
  const int  hiHalf = lane >> 4;         // lanes 16..31 -> upper K/M half
  const long row0   = (long)blockIdx.x * 16;

  // ---- cooperative staging -----------------------------------------------
  for (int i = tid; i < 16 * kT * kIn; i += 512)
    xs[i] = x[row0 * kT * kIn + i];
  for (int i = tid; i < 16 * kLdaE; i += 512)
    hA[i] = (_Float16)0.0f;
  // Streamed gates {g, o} -> LDS in B-fragment layout (extended K):
  //   fid = (slice*2 + g2)*9 + kt ; half = fid*512 + ch*256 + lane*8 + i
  for (int idx = tid; idx < kWBHalvesE; idx += 512) {
    const int fid  = idx >> 9;
    const int rem  = idx & 511;
    const int ch   = rem >> 8;
    const int l    = (rem >> 3) & 31;
    const int i    = rem & 7;
    const int kt   = fid % kKtE;
    const int rest = fid / kKtE;
    const int g2   = rest & 1;
    const int ss   = rest >> 1;
    const int col  = (2 + g2) * kH + ss * 16 + (l & 15);
    const int kg   = kt * 32 + (l >> 4) * 16 + ch * 8 + i;
    float v = 0.0f;
    if (kg < kH)                v = Whh[(long)col * kH + kg];
    else if (kg < kH + kIn)     v = Wih[col * kIn + (kg - kH)];
    else if (kg == kH + kIn)    v = bih[col] + bhh[col];     // bias column
    wB[idx] = (_Float16)v;
  }

  // ---- register-resident gates {i, f}: 9 fragments each -------------------
  v16h wf[2][kKtE];
#pragma unroll
  for (int q = 0; q < 2; ++q) {
    const int col = q * kH + j0 + nIdx;       // gates 0 (i), 1 (f)
    const float* wrow = Whh + (long)col * kH; // contiguous in K
    __builtin_prefetch(wrow, 0, 0);
#pragma unroll
    for (int kt = 0; kt < 8; ++kt) {
      const int kb = kt * 32 + hiHalf * 16;   // B layout: lane = N, 16 contig K
#pragma unroll
      for (int i2 = 0; i2 < 16; ++i2)
        wf[q][kt][i2] = (_Float16)wrow[kb + i2];
    }
    // kt = 8: x-feature columns + bias column + zero pad
#pragma unroll
    for (int i2 = 0; i2 < 16; ++i2) {
      const int kg = kH + hiHalf * 16 + i2;
      float v = 0.0f;
      if (kg < kH + kIn)          v = Wih[col * kIn + (kg - kH)];
      else if (kg == kH + kIn)    v = bih[col] + bhh[col];
      wf[q][8][i2] = (_Float16)v;
    }
  }

  v8f c;  // cell state for this lane's 8 C-layout elements
#pragma unroll
  for (int r = 0; r < 8; ++r) c[r] = 0.0f;

  __syncthreads();  // staging done (hA fully zeroed)

  // Seed the static parts of the extended A operand: x_0 and the 1.0 column.
  if (tid < 16 * kIn) {
    const int m = tid / kIn, f = tid % kIn;
    hA[m * kLdaE + kH + f] = (_Float16)xs[(m * kT + 0) * kIn + f];
  }
  if (tid < 16) hA[tid * kLdaE + kH + kIn] = (_Float16)1.0f;
  __syncthreads();

  for (int t = 0; t < kT; ++t) {
    v8f acc[4];
#pragma unroll
    for (int g = 0; g < 4; ++g)
#pragma unroll
      for (int r = 0; r < 8; ++r) acc[g][r] = 0.0f;

    // ---- acc += [h | x | 1] @ W_ext^T : 9 k-tiles x 4 gates ---------------
#pragma unroll
    for (int kt = 0; kt < kKtE; ++kt) {
      // A fragment: lane = row nIdx; K contiguous 16 (+16 for hi half)
      const _Float16* ap = &hA[nIdx * kLdaE + kt * 32 + hiHalf * 16];
      const v8h alo = *reinterpret_cast<const v8h*>(ap);
      const v8h ahi = *reinterpret_cast<const v8h*>(ap + 8);
      v16h afrag;
#pragma unroll
      for (int i2 = 0; i2 < 8; ++i2) { afrag[i2] = alo[i2]; afrag[i2 + 8] = ahi[i2]; }

      // resident gates i, f
      acc[0] = __builtin_amdgcn_wmma_f32_16x16x32_f16(
          false, afrag, false, wf[0][kt], (short)0, acc[0], false, false);
      acc[1] = __builtin_amdgcn_wmma_f32_16x16x32_f16(
          false, afrag, false, wf[1][kt], (short)0, acc[1], false, false);

      // LDS-streamed gates g, o
#pragma unroll
      for (int g2 = 0; g2 < 2; ++g2) {
        const _Float16* bp = &wB[((s * 2 + g2) * kKtE + kt) * 512 + lane * 8];
        const v8h blo = *reinterpret_cast<const v8h*>(bp);
        const v8h bhi = *reinterpret_cast<const v8h*>(bp + 256);
        v16h bfrag;
#pragma unroll
        for (int i2 = 0; i2 < 8; ++i2) { bfrag[i2] = blo[i2]; bfrag[i2 + 8] = bhi[i2]; }
        acc[2 + g2] = __builtin_amdgcn_wmma_f32_16x16x32_f16(
            false, afrag, false, bfrag, (short)0, acc[2 + g2], false, false);
      }
    }

    // ---- LSTM cell elementwise, directly on WMMA C layout -----------------
    float hval[8];
#pragma unroll
    for (int r = 0; r < 8; ++r) {
      const float gi = sigf(acc[0][r]);
      const float gf = sigf(acc[1][r]);
      const float gg = tanh_fast(acc[2][r]);
      const float go = sigf(acc[3][r]);
      const float cn = gf * c[r] + gi * gg;
      c[r]    = cn;
      hval[r] = go * tanh_fast(cn);
    }

    __syncthreads();  // all waves done reading old hA
#pragma unroll
    for (int r = 0; r < 8; ++r)
      hA[(r + 8 * hiHalf) * kLdaE + j0 + nIdx] = (_Float16)hval[r];
    // stage x_{t+1} into the extended columns for the next step
    if (t + 1 < kT && tid < 16 * kIn) {
      const int m = tid / kIn, f = tid % kIn;
      hA[m * kLdaE + kH + f] = (_Float16)xs[(m * kT + (t + 1)) * kIn + f];
    }
    __syncthreads();  // new hA visible
  }

  // Final hidden state -> global (coalesced via LDS)
  for (int i = tid; i < 16 * kH; i += 512) {
    const int m = i >> 8, j = i & 255;
    hout[(row0 + m) * kH + j] = (float)hA[m * kLdaE + j];
  }
}

// ---------------------------------------------------------------------------
// Mean over K neighbors + decoder-init projection:
//   h0 = tanh([h_t, mean_k h_n] @ init_W^T + init_b)
// Small (0.16% of FLOPs): VALU with block-level reuse of init_W rows.
// ---------------------------------------------------------------------------
__global__ __launch_bounds__(512, 1)
void combine_kernel(const float* __restrict__ ht,   // [B, H]
                    const float* __restrict__ hn,   // [B*K, H]
                    const float* __restrict__ iW,   // [H, 2H]
                    const float* __restrict__ ib,   // [H]
                    float* __restrict__ h0)         // [B, H]
{
  __shared__ float hcat[16 * 2 * kH];  // 16 rows x 512 cols, 32 KB
  const int  tid  = threadIdx.x;
  const long row0 = (long)blockIdx.x * 16;

  for (int i = tid; i < 16 * kH; i += 512) {
    const int m = i >> 8, j = i & 255;
    hcat[m * 512 + j] = ht[(row0 + m) * kH + j];
    float sum = 0.0f;
#pragma unroll
    for (int k = 0; k < kK; ++k) sum += hn[((row0 + m) * kK + k) * kH + j];
    hcat[m * 512 + kH + j] = sum * 0.125f;
  }
  __syncthreads();

  if (tid < kH) {
    const int j = tid;
    const float* wr = iW + (long)j * (2 * kH);  // one init_W row, reused x16 rows
    float a[16];
#pragma unroll
    for (int m = 0; m < 16; ++m) a[m] = ib[j];
    for (int k = 0; k < 2 * kH; ++k) {
      const float w = wr[k];
#pragma unroll
      for (int m = 0; m < 16; ++m) a[m] = fmaf(hcat[m * 512 + k], w, a[m]);
    }
#pragma unroll
    for (int m = 0; m < 16; ++m) h0[(row0 + m) * kH + j] = tanh_fast(a[m]);
  }
}

// ---------------------------------------------------------------------------
// Fused autoregressive decoder (K=256): half-resident / half-LDS WMMA
// recurrence, output head (h @ out_W^T + out_b) reduced via LDS float atomics
// and fed back through a double-buffered x buffer. Writes [B, 25, 2].
// ---------------------------------------------------------------------------
__global__ __launch_bounds__(512, 1)
void lstm_decoder_kernel(const float* __restrict__ h0,   // [B, H]
                         const float* __restrict__ Wih,  // [4H, 2]
                         const float* __restrict__ Whh,  // [4H, H]
                         const float* __restrict__ bih,  // [4H]
                         const float* __restrict__ bhh,  // [4H]
                         const float* __restrict__ outW, // [2, H]
                         const float* __restrict__ outb, // [2]
                         float* __restrict__ out)        // [B, kPred, 2]
{
  extern __shared__ char smem[];
  _Float16* wB  = (_Float16*)(smem);                          // 256 KB
  _Float16* hA  = (_Float16*)(smem + kWBBytes);               // 8448 B
  float*    sow = (float*)(smem + kWBBytes + kHABytes);       // 2048 B
  float*    xb  = (float*)(smem + kWBBytes + kHABytes + 2048);// [2][16][2]

  const int  tid    = threadIdx.x;
  const int  lane   = tid & 31;
  const int  s      = tid >> 5;
  const int  j0     = s * 16;
  const int  nIdx   = lane & 15;
  const int  hiHalf = lane >> 4;
  const long row0   = (long)blockIdx.x * 16;

  for (int i = tid; i < 16 * kH; i += 512) {
    const int m = i >> 8, j = i & 255;
    hA[m * kLda + j] = (_Float16)h0[(row0 + m) * kH + j];
  }
  for (int i = tid; i < 2 * kH; i += 512) sow[i] = outW[i];
  if (tid < 32)              { xb[0 * 32 + tid] = 0.0f; }              // x at t=0
  if (tid >= 32 && tid < 64) { xb[32 + (tid - 32)] = outb[tid & 1]; }  // accum base
  for (int idx = tid; idx < kWBHalves; idx += 512) {
    const int fid = idx >> 9;
    const int rem = idx & 511;
    const int ch  = rem >> 8;
    const int l   = (rem >> 3) & 31;
    const int i   = rem & 7;
    const int kt  = fid & 7;
    const int g2  = (fid >> 3) & 1;
    const int ss  = fid >> 4;
    const int col = (2 + g2) * kH + ss * 16 + (l & 15);
    const int kb  = kt * 32 + (l >> 4) * 16 + ch * 8 + i;
    wB[idx] = (_Float16)Whh[(long)col * kH + kb];
  }

  v16h  wf[2][8];
  float wih0[4], wih1[4], bias[4];
#pragma unroll
  for (int g = 0; g < 4; ++g) {
    const int col = g * kH + j0 + nIdx;
    bias[g] = bih[col] + bhh[col];
    wih0[g] = Wih[col * 2 + 0];
    wih1[g] = Wih[col * 2 + 1];
  }
#pragma unroll
  for (int q = 0; q < 2; ++q) {
    const int col = q * kH + j0 + nIdx;
    const float* wrow = Whh + (long)col * kH;
    __builtin_prefetch(wrow, 0, 0);
#pragma unroll
    for (int kt = 0; kt < 8; ++kt) {
      const int kb = kt * 32 + hiHalf * 16;
#pragma unroll
      for (int i2 = 0; i2 < 16; ++i2)
        wf[q][kt][i2] = (_Float16)wrow[kb + i2];
    }
  }
  v8f c;
#pragma unroll
  for (int r = 0; r < 8; ++r) c[r] = 0.0f;

  __syncthreads();

  for (int t = 0; t < kPred; ++t) {
    const int cur = t & 1, nxt = cur ^ 1;

    // ---- acc = bias + x_prev @ Wih^T (2 features: trivial VALU) ----
    v8f acc[4];
#pragma unroll
    for (int r = 0; r < 8; ++r) {
      const int   m  = r + 8 * hiHalf;
      const float x0 = xb[cur * 32 + m * 2 + 0];
      const float x1 = xb[cur * 32 + m * 2 + 1];
#pragma unroll
      for (int g = 0; g < 4; ++g)
        acc[g][r] = fmaf(x1, wih1[g], fmaf(x0, wih0[g], bias[g]));
    }

    // ---- acc += h_prev @ Whh^T (WMMA) ----
#pragma unroll
    for (int kt = 0; kt < 8; ++kt) {
      const _Float16* ap = &hA[nIdx * kLda + kt * 32 + hiHalf * 16];
      const v8h alo = *reinterpret_cast<const v8h*>(ap);
      const v8h ahi = *reinterpret_cast<const v8h*>(ap + 8);
      v16h afrag;
#pragma unroll
      for (int i2 = 0; i2 < 8; ++i2) { afrag[i2] = alo[i2]; afrag[i2 + 8] = ahi[i2]; }

      acc[0] = __builtin_amdgcn_wmma_f32_16x16x32_f16(
          false, afrag, false, wf[0][kt], (short)0, acc[0], false, false);
      acc[1] = __builtin_amdgcn_wmma_f32_16x16x32_f16(
          false, afrag, false, wf[1][kt], (short)0, acc[1], false, false);
#pragma unroll
      for (int g2 = 0; g2 < 2; ++g2) {
        const _Float16* bp = &wB[((s * 2 + g2) * 8 + kt) * 512 + lane * 8];
        const v8h blo = *reinterpret_cast<const v8h*>(bp);
        const v8h bhi = *reinterpret_cast<const v8h*>(bp + 256);
        v16h bfrag;
#pragma unroll
        for (int i2 = 0; i2 < 8; ++i2) { bfrag[i2] = blo[i2]; bfrag[i2 + 8] = bhi[i2]; }
        acc[2 + g2] = __builtin_amdgcn_wmma_f32_16x16x32_f16(
            false, afrag, false, bfrag, (short)0, acc[2 + g2], false, false);
      }
    }

    // ---- cell update ----
    float hval[8];
#pragma unroll
    for (int r = 0; r < 8; ++r) {
      const float gi = sigf(acc[0][r]);
      const float gf = sigf(acc[1][r]);
      const float gg = tanh_fast(acc[2][r]);
      const float go = sigf(acc[3][r]);
      const float cn = gf * c[r] + gi * gg;
      c[r]    = cn;
      hval[r] = go * tanh_fast(cn);
    }

    __syncthreads();  // done reading hA and xb[cur]

    // new h -> LDS; output-head partials -> xb[nxt] via ds_add_f32
    {
      const float w0 = sow[j0 + nIdx];
      const float w1 = sow[kH + j0 + nIdx];
#pragma unroll
      for (int r = 0; r < 8; ++r) {
        const int m = r + 8 * hiHalf;
        hA[m * kLda + j0 + nIdx] = (_Float16)hval[r];
        atomicAdd(&xb[nxt * 32 + m * 2 + 0], hval[r] * w0);
        atomicAdd(&xb[nxt * 32 + m * 2 + 1], hval[r] * w1);
      }
    }
    // re-arm xb[cur] (accumulation target at t+1) with out_b base
    if (tid < 32) xb[cur * 32 + tid] = outb[tid & 1];

    __syncthreads();  // prediction complete

    if (tid < 32) {
      const int m = tid >> 1, o = tid & 1;
      out[(row0 + m) * (kPred * 2) + t * 2 + o] = xb[nxt * 32 + m * 2 + o];
    }
  }
}

// ---------------------------------------------------------------------------
extern "C" void kernel_launch(void* const* d_in, const int* in_sizes, int n_in,
                              void* d_out, int out_size, void* d_ws, size_t ws_size,
                              hipStream_t stream) {
  (void)in_sizes; (void)n_in; (void)out_size; (void)ws_size;

  const float* target  = (const float*)d_in[0];   // [B, T, 7]
  const float* neigh   = (const float*)d_in[1];   // [B, K, T, 7] == [B*K, T, 7]
  // d_in[2] neigh_spatial, d_in[3] lane: unused by the reference computation
  const float* enc_Wih = (const float*)d_in[4];
  const float* enc_Whh = (const float*)d_in[5];
  const float* enc_bih = (const float*)d_in[6];
  const float* enc_bhh = (const float*)d_in[7];
  const float* nen_Wih = (const float*)d_in[8];
  const float* nen_Whh = (const float*)d_in[9];
  const float* nen_bih = (const float*)d_in[10];
  const float* nen_bhh = (const float*)d_in[11];
  const float* init_W  = (const float*)d_in[12];
  const float* init_b  = (const float*)d_in[13];
  const float* dec_Wih = (const float*)d_in[14];
  const float* dec_Whh = (const float*)d_in[15];
  const float* dec_bih = (const float*)d_in[16];
  const float* dec_bhh = (const float*)d_in[17];
  const float* out_W   = (const float*)d_in[18];
  const float* out_b   = (const float*)d_in[19];
  float* out = (float*)d_out;

  char*  ws  = (char*)d_ws;
  float* h_t = (float*)(ws);                                        // [8192, 256]
  float* h_n = (float*)(ws + (size_t)kB * kH * 4);                  // [65536, 256]
  float* h_0 = (float*)(ws + (size_t)kB * kH * 4 + (size_t)kB * kK * kH * 4);

  // Opt in to >64KB dynamic LDS (WGP supports 320KB); host-side state call,
  // safe under graph capture. Errors (if unsupported) are ignored.
  (void)hipFuncSetAttribute((const void*)lstm_encoder_kernel,
                            hipFuncAttributeMaxDynamicSharedMemorySize,
                            (int)kEncLds);
  (void)hipFuncSetAttribute((const void*)lstm_decoder_kernel,
                            hipFuncAttributeMaxDynamicSharedMemorySize,
                            (int)kDecLds);

  lstm_encoder_kernel<<<kB / 16, 512, kEncLds, stream>>>(
      target, enc_Wih, enc_Whh, enc_bih, enc_bhh, h_t);
  lstm_encoder_kernel<<<(kB * kK) / 16, 512, kEncLds, stream>>>(
      neigh, nen_Wih, nen_Whh, nen_bih, nen_bhh, h_n);
  combine_kernel<<<kB / 16, 512, 0, stream>>>(h_t, h_n, init_W, init_b, h_0);
  lstm_decoder_kernel<<<kB / 16, 512, kDecLds, stream>>>(
      h_0, dec_Wih, dec_Whh, dec_bih, dec_bhh, out_W, out_b, out);
}
